// Head_80410377715736
// MI455X (gfx1250) — matrix operand order repaired and checked by reading, not compiled, not run on hardware
//
#include <hip/hip_runtime.h>

#ifndef USE_ASYNC
#define USE_ASYNC 1     // flip to 0 if async-to-LDS mnemonics are rejected
#endif

// ---------------- types ----------------
typedef __attribute__((ext_vector_type(16))) __bf16       v16bf;
typedef __attribute__((ext_vector_type(8)))  float        v8f;
typedef __attribute__((ext_vector_type(4)))  unsigned int u32x4;
typedef __attribute__((ext_vector_type(4)))  float        f32x4;

union V16 { v16bf v; u32x4 u[2]; };

static __device__ __forceinline__ __bf16 f2bf(float f) { return (__bf16)f; }

// Problem constants: B=8, T=2048, D=1024, H=64
#define NTOK   16384   // B*T
#define TSEQ   2048
#define DDIM   1024
#define HDIM   64

#if USE_ASYNC
// one async B128 copy: 32 lanes x 16B = 512B; IOFFSET applies to LDS and global
template <int OFF>
__device__ __forceinline__ void async_cp128(unsigned lds, unsigned long long gaddr) {
  asm volatile("global_load_async_to_lds_b128 %0, %1, off offset:%2"
               :: "v"(lds), "v"(gaddr), "n"(OFF) : "memory");
}
// stage one contiguous 4KB tile (global -> LDS), 8 async issues
static __device__ __forceinline__ void stage4k(unsigned ldsByteOff, const __bf16* g, int lane) {
  unsigned lds          = ldsByteOff + (unsigned)lane * 16u;
  unsigned long long ga = (unsigned long long)(uintptr_t)g + (unsigned long long)lane * 16ull;
  async_cp128<0>(lds, ga);    async_cp128<512>(lds, ga);
  async_cp128<1024>(lds, ga); async_cp128<1536>(lds, ga);
  async_cp128<2048>(lds, ga); async_cp128<2560>(lds, ga);
  async_cp128<3072>(lds, ga); async_cp128<3584>(lds, ga);
}
#endif

// =====================================================================
// Kernel 1: pack Wq/Wk/Wv (f32 [1024,64] row-major) into bf16
// B-fragment order: [(mat*32+kt)*4+nt][lane][16 contiguous bf16],
// element e of lane l maps to k = kt*32 + (l>>4)*16 + e, n = nt*16 + (l&15)
// =====================================================================
__global__ __launch_bounds__(256) void packw_kernel(const float* __restrict__ Wq,
                                                    const float* __restrict__ Wk,
                                                    const float* __restrict__ Wv,
                                                    __bf16* __restrict__ packW) {
  int tid = blockIdx.x * 256 + threadIdx.x;     // 3*32*4*32 = 12288 total
  if (tid >= 12288) return;
  int lane = tid & 31;
  int kt   = (tid >> 7) & 31;
  int mat  = tid >> 12;
  const float* W = (mat == 0) ? Wq : (mat == 1) ? Wk : Wv;
  int n     = ((tid >> 5) & 3) * 16 + (lane & 15);
  int kbase = kt * 32 + (lane >> 4) * 16;
  __bf16* dst = packW + (size_t)tid * 16;
#pragma unroll
  for (int e = 0; e < 16; ++e) dst[e] = f2bf(W[(kbase + e) * HDIM + n]);
}

// =====================================================================
// Kernel 2: fused QKV projection via bf16 WMMA, f32 accumulate.
// One wave per (16-row tile, 16-col tile, matrix) task: 12288 tasks.
// qb/kb: bf16 row-major [NTOK,64].
// vbT  : bf16 tile-major [NTOK/32][64][32] (each 32-token V tile is a
//        contiguous 4KB block, async-stageable and B-fragment friendly).
// V fp32 goes straight to d_out (second output of the reference tuple).
// =====================================================================
__global__ __launch_bounds__(256) void qkv_kernel(const float* __restrict__ X,
                                                  const __bf16* __restrict__ packW,
                                                  __bf16* __restrict__ qb,
                                                  __bf16* __restrict__ kb,
                                                  __bf16* __restrict__ vbT,
                                                  float* __restrict__ voutF) {
  const int lane = threadIdx.x & 31;
  const int wave = threadIdx.x >> 5;
  const int task = blockIdx.x * 8 + wave;       // 12288 tasks
  if (task >= 12288) return;
  const int rowTile = task / 12;
  const int sub     = task - rowTile * 12;
  const int mat     = sub >> 2;
  const int nt      = sub & 3;
  const int row0    = rowTile * 16;
  const int m    = lane & 15;
  const int hhlf = lane >> 4;
  const int k0   = hhlf * 8;

  const float*  xrow  = X + (size_t)(row0 + m) * DDIM;
  const __bf16* wbase = packW + (((size_t)mat * 32) * 4 + nt) * 512 + (size_t)lane * 16;

  v8f acc = {};
#pragma unroll 4
  for (int kt = 0; kt < 32; ++kt) {
    // A fragment (16x32 bf16): contiguous k = 16*half + e
    const int base = kt * 32 + 2 * k0;
    f32x4 f0 = *reinterpret_cast<const f32x4*>(xrow + base);
    f32x4 f1 = *reinterpret_cast<const f32x4*>(xrow + base + 4);
    f32x4 f2 = *reinterpret_cast<const f32x4*>(xrow + base + 8);
    f32x4 f3 = *reinterpret_cast<const f32x4*>(xrow + base + 12);
    V16 a;
#pragma unroll
    for (int e = 0; e < 4; ++e) {
      a.v[e]      = f2bf(f0[e]);
      a.v[4 + e]  = f2bf(f1[e]);
      a.v[8 + e]  = f2bf(f2[e]);
      a.v[12 + e] = f2bf(f3[e]);
    }
    // B fragment: fully coalesced 32B per lane from packed weights
    const __bf16* bp = wbase + (size_t)kt * 2048;
    V16 b;
    b.u[0] = *reinterpret_cast<const u32x4*>(bp);
    b.u[1] = *reinterpret_cast<const u32x4*>(bp + 8);
    acc = __builtin_amdgcn_wmma_f32_16x16x32_bf16(false, a.v, false, b.v,
                                                  (short)0, acc, false, false);
  }

  const int n = nt * 16 + m;
#pragma unroll
  for (int j = 0; j < 8; ++j) {
    const int grow = row0 + j + 8 * hhlf;       // C layout: M = j + 8*(lane>=16)
    const float val = acc[j];
    if (mat == 0)       qb[(size_t)grow * HDIM + n] = f2bf(val);
    else if (mat == 1)  kb[(size_t)grow * HDIM + n] = f2bf(val);
    else {
      vbT[((size_t)(grow >> 5) * 64 + n) * 32 + (grow & 31)] = f2bf(val);
      voutF[(size_t)grow * HDIM + n] = val;     // exact fp32 V output
    }
  }
}

// =====================================================================
// Kernel 3: flash attention. One wave per 16-query tile; 32-key steps
// with causal early exit; online softmax in f32.
// K/V tiles double-buffered in LDS via global_load_async_to_lds_b128
// (ASYNCcnt), one full step of prefetch ahead of the WMMAs.
// P re-layout C->A goes through a per-wave LDS tile (LDS in-order per
// wave: s_wait_dscnt only, no block barrier - waves diverge in length).
// =====================================================================
__global__ __launch_bounds__(64) void attn_kernel(const __bf16* __restrict__ qb,
                                                  const __bf16* __restrict__ kb,
                                                  const __bf16* __restrict__ vbT,
                                                  float* __restrict__ outO) {
#if USE_ASYNC
  // per wave (elements): K0 @0, K1 @2048, V0 @4096, V1 @6144, P @8192
  __shared__ __align__(16) __bf16 sbuf[2][8704];   // 34,816 B total
#else
  __shared__ __align__(16) __bf16 sbuf[2][512];    // P tile only
#endif
  const int lane   = threadIdx.x & 31;
  const int wave   = threadIdx.x >> 5;
  const int tileId = blockIdx.x * 2 + wave;        // 1024 q-tiles total
  const int b  = tileId >> 7;                      // 128 tiles per batch
  const int q0 = (tileId & 127) * 16;
  const size_t tokBase = (size_t)b * TSEQ;
  const int m    = lane & 15;
  const int hhlf = lane >> 4;
  const int k0   = hhlf * 8;

  // Q fragments (A layout), H=64 -> two K=32 chunks, loaded once
  V16 qf[2];
#pragma unroll
  for (int kt = 0; kt < 2; ++kt) {
    const __bf16* qp = qb + (tokBase + q0 + m) * HDIM + kt * 32;
    qf[kt].u[0] = *reinterpret_cast<const u32x4*>(qp + 2 * k0);
    qf[kt].u[1] = *reinterpret_cast<const u32x4*>(qp + 2 * k0 + 8);
  }

  v8f acc[4];
#pragma unroll
  for (int t = 0; t < 4; ++t) acc[t] = {};
  float rmax[8], rsum[8];
#pragma unroll
  for (int j = 0; j < 8; ++j) { rmax[j] = -3.0e38f; rsum[j] = 0.0f; }

  const int nsteps = (q0 + 16 + 31) >> 5;          // causal: keys <= q0+15

#if USE_ASYNC
  const unsigned ldsBase = (unsigned)(uintptr_t)(&sbuf[wave][0]);  // LDS byte offset
  __bf16* sP = &sbuf[wave][8192];
  // prologue: stage K/V tiles for step 0 into buffer 0
  stage4k(ldsBase + 0,    kb  + (tokBase + 0) * HDIM,        lane);
  stage4k(ldsBase + 8192, vbT + ((tokBase + 0) >> 5) * 2048, lane);
#else
  __bf16* sP = &sbuf[wave][0];
#endif

  for (int stp = 0; stp < nsteps; ++stp) {
    const int s0  = stp * 32;
    const int cur = stp & 1;

#if USE_ASYNC
    if (stp + 1 < nsteps) {
      // WAR: all LDS reads of the buffer being overwritten must be done
      asm volatile("s_wait_dscnt 0x0" ::: "memory");
      const int s0n = s0 + 32;
      const int nb  = cur ^ 1;
      stage4k(ldsBase + (unsigned)nb * 4096u,         kb  + (tokBase + s0n) * HDIM,        lane);
      stage4k(ldsBase + 8192u + (unsigned)nb * 4096u, vbT + ((tokBase + s0n) >> 5) * 2048, lane);
      asm volatile("s_wait_asynccnt 0x10" ::: "memory");  // current 16 done, next 16 in flight
    } else {
      asm volatile("s_wait_asynccnt 0x0" ::: "memory");
    }
    const __bf16* sK = &sbuf[wave][(size_t)cur * 2048];
    const __bf16* sV = &sbuf[wave][4096 + (size_t)cur * 2048];
#endif

    // ---- scores S = Q K^T  (two 16x16 tiles, K-dim = 64) ----
    v8f Sv[2];
#pragma unroll
    for (int t = 0; t < 2; ++t) {
      v8f s = {};
#if USE_ASYNC
      const __bf16* kp = sK + (16 * t + m) * HDIM + hhlf * 16;
#else
      const __bf16* kp = kb + (tokBase + s0 + 16 * t + m) * HDIM + hhlf * 16;
#endif
#pragma unroll
      for (int kt = 0; kt < 2; ++kt) {
        V16 kf;
        kf.u[0] = *reinterpret_cast<const u32x4*>(kp + kt * 32);
        kf.u[1] = *reinterpret_cast<const u32x4*>(kp + kt * 32 + 8);
        s = __builtin_amdgcn_wmma_f32_16x16x32_bf16(false, qf[kt].v, false, kf.v,
                                                    (short)0, s, false, false);
      }
      Sv[t] = s;
    }

    // ---- scale + causal mask (C layout: row=q0+j+8*hhlf, col=s0+16t+m) ----
#pragma unroll
    for (int t = 0; t < 2; ++t)
#pragma unroll
      for (int j = 0; j < 8; ++j) {
        const float x   = Sv[t][j] * 0.125f;     // 1/sqrt(64)
        const int   col = s0 + 16 * t + m;
        const int   row = q0 + j + 8 * hhlf;
        Sv[t][j] = (col > row) ? -3.0e38f : x;
      }

    // ---- online softmax: row max over 32 cols (16-lane xor reduce) ----
    float mloc[8];
#pragma unroll
    for (int j = 0; j < 8; ++j) mloc[j] = fmaxf(Sv[0][j], Sv[1][j]);
#pragma unroll
    for (int off = 8; off >= 1; off >>= 1)
#pragma unroll
      for (int j = 0; j < 8; ++j)
        mloc[j] = fmaxf(mloc[j], __shfl_xor(mloc[j], off, 32));

    float alpha[8];
#pragma unroll
    for (int j = 0; j < 8; ++j) {
      const float nm = fmaxf(rmax[j], mloc[j]);
      alpha[j] = __expf(rmax[j] - nm);
      rmax[j]  = nm;
    }
#pragma unroll
    for (int t = 0; t < 4; ++t)
#pragma unroll
      for (int j = 0; j < 8; ++j) acc[t][j] *= alpha[j];

    float pv0[8], pv1[8], ps[8];
#pragma unroll
    for (int j = 0; j < 8; ++j) {
      pv0[j] = __expf(Sv[0][j] - rmax[j]);
      pv1[j] = __expf(Sv[1][j] - rmax[j]);
      ps[j]  = pv0[j] + pv1[j];
    }
#pragma unroll
    for (int off = 8; off >= 1; off >>= 1)
#pragma unroll
      for (int j = 0; j < 8; ++j) ps[j] += __shfl_xor(ps[j], off, 32);
#pragma unroll
    for (int j = 0; j < 8; ++j) rsum[j] = rsum[j] * alpha[j] + ps[j];

    // ---- P (C layout) -> LDS row-major [16][32] bf16 -> A fragment ----
#pragma unroll
    for (int j = 0; j < 8; ++j) {
      const int r = j + 8 * hhlf;
      sP[r * 32 + m]      = f2bf(pv0[j]);
      sP[r * 32 + 16 + m] = f2bf(pv1[j]);
    }
    asm volatile("s_wait_dscnt 0x0" ::: "memory");   // LDS in-order per wave
    V16 pa;
    pa.u[0] = *reinterpret_cast<const u32x4*>(sP + m * 32 + 2 * k0);
    pa.u[1] = *reinterpret_cast<const u32x4*>(sP + m * 32 + 2 * k0 + 8);

    // ---- acc += P @ V ----
#pragma unroll
    for (int ht = 0; ht < 4; ++ht) {
#if USE_ASYNC
      const __bf16* vp = sV + (ht * 16 + m) * 32 + hhlf * 16;
#else
      const __bf16* vp = vbT + (((tokBase + s0) >> 5) * 64 + ht * 16 + m) * 32 + hhlf * 16;
#endif
      V16 vf;
      vf.u[0] = *reinterpret_cast<const u32x4*>(vp);
      vf.u[1] = *reinterpret_cast<const u32x4*>(vp + 8);
      acc[ht] = __builtin_amdgcn_wmma_f32_16x16x32_bf16(false, pa.v, false, vf.v,
                                                        (short)0, acc[ht], false, false);
    }
    asm volatile("" ::: "memory");
  }

  // ---- epilogue: normalize and store fp32 ----
#pragma unroll
  for (int ht = 0; ht < 4; ++ht)
#pragma unroll
    for (int j = 0; j < 8; ++j) {
      const int r = j + 8 * hhlf;
      outO[(tokBase + q0 + r) * HDIM + ht * 16 + m] = acc[ht][j] / rsum[j];
    }
}

// =====================================================================
// Launch
// =====================================================================
extern "C" void kernel_launch(void* const* d_in, const int* in_sizes, int n_in,
                              void* d_out, int out_size, void* d_ws, size_t ws_size,
                              hipStream_t stream) {
  const float* X  = (const float*)d_in[0];   // [8,2048,1024]
  const float* Wq = (const float*)d_in[1];   // [1024,64]
  const float* Wk = (const float*)d_in[2];
  const float* Wv = (const float*)d_in[3];

  float* outO = (float*)d_out;                              // out [B,T,H]
  float* outV = (float*)d_out + (size_t)NTOK * HDIM;        // v   [B,T,H]

  char* ws = (char*)d_ws;
  __bf16* packW = (__bf16*)(ws);                            // 384 KB
  __bf16* qb    = (__bf16*)(ws + 393216);                   // 2 MB
  __bf16* kb    = (__bf16*)(ws + 393216 + 2097152);         // 2 MB
  __bf16* vbT   = (__bf16*)(ws + 393216 + 2 * 2097152);     // 2 MB

  hipLaunchKernelGGL(packw_kernel, dim3(48),   dim3(256), 0, stream, Wq, Wk, Wv, packW);
  hipLaunchKernelGGL(qkv_kernel,   dim3(1536), dim3(256), 0, stream, X, packW, qb, kb, vbT, outV);
  hipLaunchKernelGGL(attn_kernel,  dim3(512),  dim3(64),  0, stream, qb, kb, vbT, outO);
}
